// SAVANet_17317308137729
// MI455X (gfx1250) — matrix operand order repaired and checked by reading, not compiled, not run on hardware
//
#include <hip/hip_runtime.h>
#include <hip/hip_bf16.h>
#include <math.h>

// ---------------------------------------------------------------------------
// SAVANet forward for MI455X (gfx1250, wave32, WMMA).
// Big GEMMs: v_wmma_f32_16x16x32_bf16 (f32 accumulate), LDS-tiled 64x64 blocks.
// Exactness-sensitive GEMMs (cov, Newton-Schulz, 1x1 convs): v_wmma_f32_16x16x4_f32.
// ZCA eigh replaced by trace-normalized Newton-Schulz inverse-sqrt (GEMM only).
// Staging uses GLOBAL_LOAD_ASYNC_TO_LDS_B128 + s_wait_asynccnt where the LDS
// destination is lane-contiguous (guarded by __has_builtin), packed-pair
// transpose stores elsewhere, and global_prefetch for the next K tile.
// ---------------------------------------------------------------------------

typedef __attribute__((ext_vector_type(2)))  float  v2f;
typedef __attribute__((ext_vector_type(4)))  float  f32x4;
typedef __attribute__((ext_vector_type(8)))  float  v8f;
typedef __attribute__((ext_vector_type(16))) __bf16 v16bf;
typedef __attribute__((ext_vector_type(8)))  unsigned short u16x8;
typedef __attribute__((ext_vector_type(4)))  unsigned short u16x4;
typedef __attribute__((ext_vector_type(4)))  int    i32x4;

#define B_   2
#define C_   512
#define N_   4096               // H*W
#define CHW  ((long)C_ * N_)    // 2097152
#define BCHW ((long)B_ * CHW)   // 4194304
#define N2   ((long)N_ * N_)    // 16777216
#define KEEP_K 3788             // int(4096*0.925)
#define NS_ITERS 6

#if __has_builtin(__builtin_amdgcn_global_load_async_to_lds_b128) && \
    __has_builtin(__builtin_amdgcn_s_wait_asynccnt)
#define HAVE_ASYNC_LDS 1
#define AS1_ __attribute__((address_space(1)))
#define AS3_ __attribute__((address_space(3)))
__device__ __forceinline__ void async_cp16(const void* g, void* l) {
  __builtin_amdgcn_global_load_async_to_lds_b128(
      (AS1_ i32x4*)(unsigned long long)g, (AS3_ i32x4*)l, 0, 0);
}
__device__ __forceinline__ void async_fence() {
  __builtin_amdgcn_s_wait_asynccnt(0);
}
#else
#define HAVE_ASYNC_LDS 0
__device__ __forceinline__ void async_fence() {}
#endif

__device__ __forceinline__ unsigned short f2bf(float f) {
  union { float f; unsigned u; } x; x.f = f;
  unsigned r = x.u + 0x7FFFu + ((x.u >> 16) & 1u);   // round-to-nearest-even
  return (unsigned short)(r >> 16);
}

// ---------------------------------------------------------------------------
// f32 WMMA GEMM, LDS tiled.  D = alpha*A*B (+bias)(+eps*I)(+add)
// Block: 256 threads (8 waves), 64x64 D tile, K staged in steps of 16.
// LDS row pitch 20 floats (80B) -> 16B-aligned rows for async B128 stores.
// Requires: M,N multiples of 64; K multiple of 16; one of saR/saC==1, one of
// sbR/sbC==1 (all call sites satisfy this).
// ---------------------------------------------------------------------------
#define F32_TK 16
#define F32_PAD 4
__global__ __launch_bounds__(256)
void gemm_f32_wmma(const float* __restrict__ A, long saR, long saC, long saB,
                   const float* __restrict__ Bm, long sbR, long sbC, long sbB,
                   float* __restrict__ D, long ldD, long sdB,
                   const float* __restrict__ bias,
                   const float* __restrict__ add, long sAddB,
                   float alpha, float diag_eps, int M, int N, int K) {
  __shared__ float Asf[64][F32_TK + F32_PAD];   // [m][k]
  __shared__ float Bsf[64][F32_TK + F32_PAD];   // [n][k]
  const int tid  = threadIdx.x;
  const int wave = tid >> 5;
  const int lane = tid & 31;
  const int half = lane >> 4;
  const int mr   = lane & 15;
  const int m0 = blockIdx.y * 64, n0 = blockIdx.x * 64, b = blockIdx.z;
  A  += (long)b * saB;
  Bm += (long)b * sbB;
  const int wm = wave & 3;          // row 16-block within 64
  const int wn = wave >> 2;         // col 32-block within 64 (2 subtiles)
  v8f acc0 = {}, acc1 = {};

  for (int k0 = 0; k0 < K; k0 += F32_TK) {
    // ---- stage A tile (64 x 16): all call sites have saC==1 ----
    if (saC == 1) {                 // unit stride along k -> contiguous 16B
      const int m = tid >> 2, kc = (tid & 3) * 4;
      const float* gp = &A[(long)(m0 + m) * saR + (k0 + kc)];
#if HAVE_ASYNC_LDS
      async_cp16(gp, &Asf[m][kc]);
#else
      *(f32x4*)&Asf[m][kc] = *(const f32x4*)gp;
#endif
      if (k0 + F32_TK < K) __builtin_prefetch(gp + F32_TK, 0, 1);
    } else {                        // saR == 1: packed-pair transpose store
      const int mc = (tid & 31) * 2, k2 = (tid >> 5) * 2;
      const float* gp0 = &A[(long)(k0 + k2) * saC + (m0 + mc)];
      v2f r0 = *(const v2f*)gp0;
      v2f r1 = *(const v2f*)(gp0 + saC);
      v2f p0; p0.x = r0.x; p0.y = r1.x;
      v2f p1; p1.x = r0.y; p1.y = r1.y;
      *(v2f*)&Asf[mc][k2]     = p0;
      *(v2f*)&Asf[mc + 1][k2] = p1;
      if (k0 + F32_TK < K) __builtin_prefetch(gp0 + F32_TK * saC, 0, 1);
    }
    // ---- stage B tile (16 x 64) into [n][k] ----
    if (sbC == 1) {                 // unit stride along n: packed-pair transpose
      const int nc = (tid & 31) * 2, k2 = (tid >> 5) * 2;
      const float* gp0 = &Bm[(long)(k0 + k2) * sbR + (n0 + nc)];
      v2f r0 = *(const v2f*)gp0;
      v2f r1 = *(const v2f*)(gp0 + sbR);
      v2f p0; p0.x = r0.x; p0.y = r1.x;
      v2f p1; p1.x = r0.y; p1.y = r1.y;
      *(v2f*)&Bsf[nc][k2]     = p0;
      *(v2f*)&Bsf[nc + 1][k2] = p1;
      if (k0 + F32_TK < K) __builtin_prefetch(gp0 + F32_TK * sbR, 0, 1);
    } else {                        // sbR == 1: unit stride along k -> contiguous
      const int n = tid >> 2, kc = (tid & 3) * 4;
      const float* gp = &Bm[(long)(n0 + n) * sbC + (k0 + kc)];
#if HAVE_ASYNC_LDS
      async_cp16(gp, &Bsf[n][kc]);
#else
      *(f32x4*)&Bsf[n][kc] = *(const f32x4*)gp;
#endif
      if (k0 + F32_TK < K) __builtin_prefetch(gp + F32_TK, 0, 1);
    }
#if HAVE_ASYNC_LDS
    if (saC == 1 || sbR == 1) async_fence();
#endif
    __syncthreads();
    // ---- 4 WMMA K-steps of 4 over the staged tile ----
    const int arow = wm * 16 + mr;
    const int brow0 = wn * 32 + mr;
#pragma unroll
    for (int kk = 0; kk < F32_TK; kk += 4) {
      const int ka = kk + 2 * half;
      v2f a  = *(const v2f*)&Asf[arow][ka];
      v2f b0 = *(const v2f*)&Bsf[brow0][ka];
      v2f b1 = *(const v2f*)&Bsf[brow0 + 16][ka];
      acc0 = __builtin_amdgcn_wmma_f32_16x16x4_f32(false, a, false, b0,
                                                   (short)0, acc0, false, false);
      acc1 = __builtin_amdgcn_wmma_f32_16x16x4_f32(false, a, false, b1,
                                                   (short)0, acc1, false, false);
    }
    __syncthreads();
  }

  float* Db = D + (long)b * sdB;
  const float* addb = add ? (add + (long)b * sAddB) : nullptr;
  const int col0 = n0 + wn * 32 + mr;
#pragma unroll
  for (int r = 0; r < 8; ++r) {
    const int row = m0 + wm * 16 + r + 8 * half;
    float v0 = acc0[r] * alpha;
    float v1 = acc1[r] * alpha;
    if (bias) { v0 += bias[row]; v1 += bias[row]; }
    if (diag_eps != 0.0f) {
      if (row == col0)      v0 += diag_eps;
      if (row == col0 + 16) v1 += diag_eps;
    }
    if (addb) {
      v0 += addb[(long)row * ldD + col0];
      v1 += addb[(long)row * ldD + col0 + 16];
    }
    Db[(long)row * ldD + col0]      = v0;
    Db[(long)row * ldD + col0 + 16] = v1;
  }
}

// ---------------------------------------------------------------------------
// bf16 WMMA GEMM (f32 accumulate), LDS tiled.  Block: 256 threads, 64x64 tile,
// K staged in steps of 32 (one wmma K).  LDS row pitch 40 ushorts (80B) ->
// 16B-aligned rows for async B128 stores; packed (K,K+1) dwords everywhere.
// Fragment layouts used here (contiguous k order, same for A and B):
//   A 16x32: lane half h: VGPR j -> K=2j+16h, row = lane&15
//   B 32x16: VGPR j -> K=2j+16h, col = lane&15
// ---------------------------------------------------------------------------
#define BF_TK 32
#define BF_PAD 8
__global__ __launch_bounds__(256)
void gemm_bf16_wmma(const unsigned short* __restrict__ A, long saR, long saC, long saB,
                    const unsigned short* __restrict__ Bm, long sbR, long sbC, long sbB,
                    float* __restrict__ D, long ldD, long sdB,
                    const float* __restrict__ bias, float alpha,
                    int M, int N, int K) {
  __shared__ unsigned short As16[64][BF_TK + BF_PAD];  // [m][k]
  __shared__ unsigned short Bs16[64][BF_TK + BF_PAD];  // [n][k]
  const int tid  = threadIdx.x;
  const int wave = tid >> 5;
  const int lane = tid & 31;
  const int half = lane >> 4;
  const int mr   = lane & 15;
  const int m0 = blockIdx.y * 64, n0 = blockIdx.x * 64, b = blockIdx.z;
  A  += (long)b * saB;
  Bm += (long)b * sbB;
  const int wm = wave & 3;
  const int wn = wave >> 2;
  v8f acc0 = {}, acc1 = {};

  for (int k0 = 0; k0 < K; k0 += BF_TK) {
    // ---- stage A tile (64 x 32) ----
    if (saC == 1) {                 // unit stride along k -> contiguous 16B
      const int m = tid >> 2, kc = (tid & 3) * 8;
      const unsigned short* gp = &A[(long)(m0 + m) * saR + (k0 + kc)];
#if HAVE_ASYNC_LDS
      async_cp16(gp, &As16[m][kc]);
#else
      *(u16x8*)&As16[m][kc] = *(const u16x8*)gp;
#endif
      if (k0 + BF_TK < K) __builtin_prefetch(gp + BF_TK, 0, 1);
    } else {                        // saR == 1: packed-pair transpose store
      const int mc = (tid & 15) * 4, k2 = (tid >> 4) * 2;
      const unsigned short* gp0 = &A[(long)(k0 + k2) * saC + (m0 + mc)];
      u16x4 r0 = *(const u16x4*)gp0;
      u16x4 r1 = *(const u16x4*)(gp0 + saC);
#pragma unroll
      for (int e = 0; e < 4; ++e)
        *(unsigned*)&As16[mc + e][k2] = (unsigned)r0[e] | ((unsigned)r1[e] << 16);
      if (k0 + BF_TK < K) __builtin_prefetch(gp0 + BF_TK * saC, 0, 1);
    }
    // ---- stage B tile (32 x 64) into [n][k] ----
    if (sbC == 1) {                 // unit stride along n: packed-pair transpose
      const int nc = (tid & 15) * 4, k2 = (tid >> 4) * 2;
      const unsigned short* gp0 = &Bm[(long)(k0 + k2) * sbR + (n0 + nc)];
      u16x4 r0 = *(const u16x4*)gp0;
      u16x4 r1 = *(const u16x4*)(gp0 + sbR);
#pragma unroll
      for (int e = 0; e < 4; ++e)
        *(unsigned*)&Bs16[nc + e][k2] = (unsigned)r0[e] | ((unsigned)r1[e] << 16);
      if (k0 + BF_TK < K) __builtin_prefetch(gp0 + BF_TK * sbR, 0, 1);
    } else {                        // sbR == 1: unit stride along k -> contiguous
      const int n = tid >> 2, kc = (tid & 3) * 8;
      const unsigned short* gp = &Bm[(long)(n0 + n) * sbC + (k0 + kc)];
#if HAVE_ASYNC_LDS
      async_cp16(gp, &Bs16[n][kc]);
#else
      *(u16x8*)&Bs16[n][kc] = *(const u16x8*)gp;
#endif
      if (k0 + BF_TK < K) __builtin_prefetch(gp + BF_TK, 0, 1);
    }
#if HAVE_ASYNC_LDS
    if (saC == 1 || sbR == 1) async_fence();
#endif
    __syncthreads();
    // ---- build fragments (packed pair reads) + 2 WMMAs ----
    union { unsigned u[8]; v16bf v; } av, bv0, bv1;
    const int arow = wm * 16 + mr;
    const int brow = wn * 32 + mr;
#pragma unroll
    for (int j = 0; j < 8; ++j) {
      const int ka = 2 * j + 16 * half;
      av.u[j] = *(const unsigned*)&As16[arow][ka];
    }
#pragma unroll
    for (int j = 0; j < 8; ++j) {
      const int kb = 2 * j + 16 * half;
      bv0.u[j] = *(const unsigned*)&Bs16[brow][kb];
      bv1.u[j] = *(const unsigned*)&Bs16[brow + 16][kb];
    }
    acc0 = __builtin_amdgcn_wmma_f32_16x16x32_bf16(false, av.v, false, bv0.v,
                                                   (short)0, acc0, false, false);
    acc1 = __builtin_amdgcn_wmma_f32_16x16x32_bf16(false, av.v, false, bv1.v,
                                                   (short)0, acc1, false, false);
    __syncthreads();
  }

  float* Db = D + (long)b * sdB;
  const int col0 = n0 + wn * 32 + mr;
#pragma unroll
  for (int r = 0; r < 8; ++r) {
    const int row = m0 + wm * 16 + r + 8 * half;
    float v0 = acc0[r] * alpha;
    float v1 = acc1[r] * alpha;
    if (bias) { v0 += bias[row]; v1 += bias[row]; }
    Db[(long)row * ldD + col0]      = v0;
    Db[(long)row * ldD + col0 + 16] = v1;
  }
}

// ---------------------------------------------------------------------------
// Per-(b,c)-row normalization over n elements. mode 1: AdaIN, mode 0: center.
// ---------------------------------------------------------------------------
__global__ __launch_bounds__(256)
void rownorm_kernel(const float* __restrict__ src, float* __restrict__ dst,
                    int n, int mode) {
  const long base = (long)blockIdx.x * n;
  __shared__ float s1[256], s2[256];
  float sum = 0.f, sq = 0.f;
  for (int j = threadIdx.x; j < n; j += 256) {
    float v = src[base + j]; sum += v; sq += v * v;
  }
  s1[threadIdx.x] = sum; s2[threadIdx.x] = sq;
  __syncthreads();
  for (int s = 128; s > 0; s >>= 1) {
    if (threadIdx.x < s) { s1[threadIdx.x] += s1[threadIdx.x + s];
                           s2[threadIdx.x] += s2[threadIdx.x + s]; }
    __syncthreads();
  }
  const float m   = s1[0] / n;
  const float var = s2[0] / n - m * m;
  const float inv = (mode == 1) ? rsqrtf(var + 1e-5f) : 1.0f;
  for (int j = threadIdx.x; j < n; j += 256)
    dst[base + j] = (src[base + j] - m) * inv;
}

__global__ __launch_bounds__(256)
void tobf16_kernel(const float* __restrict__ src, unsigned short* __restrict__ dst, long n) {
  long i = (long)blockIdx.x * 256 + threadIdx.x;
  if (i < n) dst[i] = f2bf(src[i]);
}

// In-place row softmax over n columns; one block per row.
__global__ __launch_bounds__(256)
void softmax_rows_kernel(float* __restrict__ mat, int n) {
  const long base = (long)blockIdx.x * n;
  __shared__ float red[256];
  float mx = -3.4e38f;
  for (int j = threadIdx.x; j < n; j += 256) mx = fmaxf(mx, mat[base + j]);
  red[threadIdx.x] = mx; __syncthreads();
  for (int s = 128; s > 0; s >>= 1) {
    if (threadIdx.x < s) red[threadIdx.x] = fmaxf(red[threadIdx.x], red[threadIdx.x + s]);
    __syncthreads();
  }
  mx = red[0]; __syncthreads();
  float sum = 0.f;
  for (int j = threadIdx.x; j < n; j += 256) {
    float e = __expf(mat[base + j] - mx);
    mat[base + j] = e; sum += e;
  }
  red[threadIdx.x] = sum; __syncthreads();
  for (int s = 128; s > 0; s >>= 1) {
    if (threadIdx.x < s) red[threadIdx.x] += red[threadIdx.x + s];
    __syncthreads();
  }
  const float inv = 1.0f / red[0];
  for (int j = threadIdx.x; j < n; j += 256) mat[base + j] *= inv;
}

// out[j] = scale * sum_i mat[i*n + j]
__global__ __launch_bounds__(256)
void colmean_kernel(const float* __restrict__ mat, int rows, int n,
                    float* __restrict__ out, float scale) {
  int j = blockIdx.x * 256 + threadIdx.x;
  if (j >= n) return;
  float s = 0.f;
  for (int i = 0; i < rows; ++i) s += mat[(long)i * n + j];
  out[j] = s * scale;
}

__global__ __launch_bounds__(256)
void trace_kernel(const float* __restrict__ A, int c, float* __restrict__ tr) {
  __shared__ float red[256];
  float s = 0.f;
  for (int i = threadIdx.x; i < c; i += 256) s += A[(long)i * c + i];
  red[threadIdx.x] = s; __syncthreads();
  for (int st = 128; st > 0; st >>= 1) {
    if (threadIdx.x < st) red[threadIdx.x] += red[threadIdx.x + st];
    __syncthreads();
  }
  if (threadIdx.x == 0) tr[0] = red[0];
}

// Y = cov / tr, Z = I
__global__ __launch_bounds__(256)
void ns_init_kernel(const float* __restrict__ cov, float* __restrict__ Y,
                    float* __restrict__ Z, const float* __restrict__ tr, int c) {
  long i = (long)blockIdx.x * 256 + threadIdx.x;
  long tot = (long)c * c;
  if (i >= tot) return;
  float invtr = 1.0f / tr[0];
  Y[i] = cov[i] * invtr;
  Z[i] = ((i / c) == (i % c)) ? 1.0f : 0.0f;
}

// P = 0.5*(3I - P)
__global__ __launch_bounds__(256)
void ns_t_kernel(float* __restrict__ P, int c) {
  long i = (long)blockIdx.x * 256 + threadIdx.x;
  long tot = (long)c * c;
  if (i >= tot) return;
  float d = ((i / c) == (i % c)) ? 3.0f : 0.0f;
  P[i] = 0.5f * (d - P[i]);
}

// Z *= rsqrt(tr)   (whiten = (cov/tr)^{-1/2} / sqrt(tr))
__global__ __launch_bounds__(256)
void ns_fin_kernel(float* __restrict__ Z, const float* __restrict__ tr, int c) {
  long i = (long)blockIdx.x * 256 + threadIdx.x;
  long tot = (long)c * c;
  if (i >= tot) return;
  Z[i] *= rsqrtf(tr[0]);
}

// keep[j] = 1 iff rank of atten[b][j] (stable ascending) >= k for BOTH batches.
__global__ __launch_bounds__(256)
void keepmask_kernel(const float* __restrict__ atten, int n, int k,
                     float* __restrict__ keep) {
  __shared__ float a0[N_];
  __shared__ float a1[N_];
  for (int j = threadIdx.x; j < n; j += 256) { a0[j] = atten[j]; a1[j] = atten[n + j]; }
  __syncthreads();
  for (int j = blockIdx.x * 256 + threadIdx.x; j < n; j += gridDim.x * 256) {
    const float v0 = a0[j], v1 = a1[j];
    int r0 = 0, r1 = 0;
    for (int t = 0; t < n; ++t) {
      const float u0 = a0[t], u1 = a1[t];
      r0 += (u0 < v0) || (u0 == v0 && t < j);
      r1 += (u1 < v1) || (u1 == v1 && t < j);
    }
    keep[j] = (r0 >= k && r1 >= k) ? 1.0f : 0.0f;
  }
}

// corrb16[row,:] = bf16( (keepx[i]==keepy[j]) * corr[row,:] / max(l1,1e-12) )
__global__ __launch_bounds__(256)
void mask_renorm_kernel(const float* __restrict__ corr,
                        const float* __restrict__ keepx,
                        const float* __restrict__ keepy,
                        unsigned short* __restrict__ corrb, int n) {
  const long row = blockIdx.x;           // 0 .. B*n-1
  const int  i   = (int)(row % n);
  const long base = row * n;
  const float kx = keepx[i];
  __shared__ float red[256];
  float s = 0.f;
  for (int j = threadIdx.x; j < n; j += 256) {
    float m = (kx == keepy[j]) ? 1.0f : 0.0f;
    s += fabsf(corr[base + j] * m);
  }
  red[threadIdx.x] = s; __syncthreads();
  for (int st = 128; st > 0; st >>= 1) {
    if (threadIdx.x < st) red[threadIdx.x] += red[threadIdx.x + st];
    __syncthreads();
  }
  const float inv = 1.0f / fmaxf(red[0], 1e-12f);
  for (int j = threadIdx.x; j < n; j += 256) {
    float m = (kx == keepy[j]) ? 1.0f : 0.0f;
    corrb[base + j] = f2bf(corr[base + j] * m * inv);
  }
}

__global__ __launch_bounds__(256)
void copy_kernel(const float* __restrict__ src, float* __restrict__ dst, long n) {
  long i = (long)blockIdx.x * 256 + threadIdx.x;
  if (i < n) dst[i] = src[i];
}

// ---------------------------------------------------------------------------
// Host side
// ---------------------------------------------------------------------------
static inline void gemmF32(hipStream_t s,
                           const float* A, long saR, long saC, long saB,
                           const float* Bm, long sbR, long sbC, long sbB,
                           float* D, long ldD, long sdB,
                           const float* bias, const float* add, long sAddB,
                           float alpha, float diag_eps,
                           int M, int N, int K, int batches) {
  dim3 g(N / 64, M / 64, batches);
  gemm_f32_wmma<<<g, 256, 0, s>>>(A, saR, saC, saB, Bm, sbR, sbC, sbB,
                                  D, ldD, sdB, bias, add, sAddB,
                                  alpha, diag_eps, M, N, K);
}
static inline void gemmBF16(hipStream_t s,
                            const unsigned short* A, long saR, long saC, long saB,
                            const unsigned short* Bm, long sbR, long sbC, long sbB,
                            float* D, long ldD, long sdB,
                            const float* bias, float alpha,
                            int M, int N, int K, int batches) {
  dim3 g(N / 64, M / 64, batches);
  gemm_bf16_wmma<<<g, 256, 0, s>>>(A, saR, saC, saB, Bm, sbR, sbC, sbB,
                                   D, ldD, sdB, bias, alpha, M, N, K);
}

extern "C" void kernel_launch(void* const* d_in, const int* in_sizes, int n_in,
                              void* d_out, int out_size, void* d_ws, size_t ws_size,
                              hipStream_t stream) {
  const float* x     = (const float*)d_in[0];
  const float* y     = (const float*)d_in[1];
  const float* f_w   = (const float*)d_in[2];
  const float* f_b   = (const float*)d_in[3];
  const float* g_w   = (const float*)d_in[4];
  const float* g_b   = (const float*)d_in[5];
  const float* sa_f_w = (const float*)d_in[6];
  const float* sa_f_b = (const float*)d_in[7];
  const float* sa_g_w = (const float*)d_in[8];
  const float* sa_g_b = (const float*)d_in[9];
  const float* h_w   = (const float*)d_in[10];
  const float* h_b   = (const float*)d_in[11];
  const float* out_w = (const float*)d_in[12];
  const float* out_b = (const float*)d_in[13];
  float* out = (float*)d_out;

  char* ws = (char*)d_ws;
  size_t o = 0;
  float* corr = (float*)(ws + o);           o += (size_t)B_ * N2 * 4;   // 134 MB
  // esc (f32, N2) aliases corrb16 (ushort, B_*N2): identical 67 MB footprint,
  // disjoint lifetimes (esc used for self-attn energies, corrb written later).
  float*          esc   = (float*)(ws + o);
  unsigned short* corrb = (unsigned short*)(ws + o); o += (size_t)N2 * 4;
  float* xz   = (float*)(ws + o); o += (size_t)BCHW * 4;
  float* yz   = (float*)(ws + o); o += (size_t)BCHW * 4;
  float* fbuf = (float*)(ws + o); o += (size_t)BCHW * 4;
  float* gbuf = (float*)(ws + o); o += (size_t)BCHW * 4;
  unsigned short* fb16 = (unsigned short*)(ws + o); o += (size_t)BCHW * 2;
  unsigned short* gb16 = (unsigned short*)(ws + o); o += (size_t)BCHW * 2;
  float* attenx = (float*)(ws + o); o += (size_t)B_ * N_ * 4;
  float* atteny = (float*)(ws + o); o += (size_t)B_ * N_ * 4;
  float* keepx  = (float*)(ws + o); o += (size_t)N_ * 4;
  float* keepy  = (float*)(ws + o); o += (size_t)N_ * 4;
  const size_t cc = (size_t)C_ * C_;
  float* cov = (float*)(ws + o); o += cc * 4;
  float* Y   = (float*)(ws + o); o += cc * 4;
  float* Z   = (float*)(ws + o); o += cc * 4;
  float* Y2  = (float*)(ws + o); o += cc * 4;
  float* Z2  = (float*)(ws + o); o += cc * 4;
  float* P   = (float*)(ws + o); o += cc * 4;
  float* tr  = (float*)(ws + o); o += 256;
  (void)ws_size; (void)in_sizes; (void)n_in; (void)out_size;

  const long ccl = (long)cc;
  const int bcBlocks = B_ * C_;

  // ---- AdaIN(x)->xz (temp), AdaIN(y)->yz (temp); f = Wf*xa+bf, g = Wg*ya+bg
  rownorm_kernel<<<bcBlocks, 256, 0, stream>>>(x, xz, N_, 1);
  rownorm_kernel<<<bcBlocks, 256, 0, stream>>>(y, yz, N_, 1);
  gemmF32(stream, f_w, C_, 1, 0, xz, N_, 1, CHW, fbuf, N_, CHW,
          f_b, nullptr, 0, 1.0f, 0.0f, C_, N_, C_, B_);
  gemmF32(stream, g_w, C_, 1, 0, yz, N_, 1, CHW, gbuf, N_, CHW,
          g_b, nullptr, 0, 1.0f, 0.0f, C_, N_, C_, B_);
  {
    long tot = BCHW;
    tobf16_kernel<<<(tot + 255) / 256, 256, 0, stream>>>(fbuf, fb16, tot);
    tobf16_kernel<<<(tot + 255) / 256, 256, 0, stream>>>(gbuf, gb16, tot);
  }
  // ---- main correlation energy E[b,i,j] = sum_c f[b,c,i] g[b,c,j]; softmax rows
  gemmBF16(stream, fb16, 1, N_, CHW, gb16, N_, 1, CHW, corr, N_, N2,
           nullptr, 1.0f, N_, N_, C_, B_);
  softmax_rows_kernel<<<B_ * N_, 256, 0, stream>>>(corr, N_);

  // ---- ZCA whitening via Newton-Schulz (per tensor, per batch)
  const float* srcs[2] = { x, y };
  float*       dsts[2] = { xz, yz };
  for (int t = 0; t < 2; ++t) {
    for (int b = 0; b < B_; ++b) {
      const float* src = srcs[t] + (long)b * CHW;
      float*       dst = dsts[t] + (long)b * CHW;
      rownorm_kernel<<<C_, 256, 0, stream>>>(src, fbuf, N_, 0);      // fc
      gemmF32(stream, fbuf, N_, 1, 0, fbuf, 1, N_, 0, cov, C_, 0,
              nullptr, nullptr, 0, 1.0f / (N_ - 1), 1e-5f, C_, C_, N_, 1);
      trace_kernel<<<1, 256, 0, stream>>>(cov, C_, tr);
      ns_init_kernel<<<(int)((ccl + 255) / 256), 256, 0, stream>>>(cov, Y, Z, tr, C_);
      float *Yc = Y, *Zc = Z, *Yn = Y2, *Zn = Z2;
      for (int it = 0; it < NS_ITERS; ++it) {
        gemmF32(stream, Zc, C_, 1, 0, Yc, C_, 1, 0, P, C_, 0,
                nullptr, nullptr, 0, 1.0f, 0.0f, C_, C_, C_, 1);
        ns_t_kernel<<<(int)((ccl + 255) / 256), 256, 0, stream>>>(P, C_);
        gemmF32(stream, Yc, C_, 1, 0, P, C_, 1, 0, Yn, C_, 0,
                nullptr, nullptr, 0, 1.0f, 0.0f, C_, C_, C_, 1);
        gemmF32(stream, P, C_, 1, 0, Zc, C_, 1, 0, Zn, C_, 0,
                nullptr, nullptr, 0, 1.0f, 0.0f, C_, C_, C_, 1);
        float* tswap;
        tswap = Yc; Yc = Yn; Yn = tswap;
        tswap = Zc; Zc = Zn; Zn = tswap;
      }
      ns_fin_kernel<<<(int)((ccl + 255) / 256), 256, 0, stream>>>(Zc, tr, C_);
      gemmF32(stream, Zc, C_, 1, 0, fbuf, N_, 1, 0, dst, N_, 0,
              nullptr, nullptr, 0, 1.0f, 0.0f, C_, N_, C_, 1);
    }
  }

  // ---- self-attention attens on xz and yz (only column means kept)
  float* zsrcs[2]  = { xz, yz };
  float* attens[2] = { attenx, atteny };
  for (int t = 0; t < 2; ++t) {
    gemmF32(stream, sa_f_w, C_, 1, 0, zsrcs[t], N_, 1, CHW, fbuf, N_, CHW,
            sa_f_b, nullptr, 0, 1.0f, 0.0f, C_, N_, C_, B_);
    gemmF32(stream, sa_g_w, C_, 1, 0, zsrcs[t], N_, 1, CHW, gbuf, N_, CHW,
            sa_g_b, nullptr, 0, 1.0f, 0.0f, C_, N_, C_, B_);
    tobf16_kernel<<<(int)((BCHW + 255) / 256), 256, 0, stream>>>(fbuf, fb16, BCHW);
    tobf16_kernel<<<(int)((BCHW + 255) / 256), 256, 0, stream>>>(gbuf, gb16, BCHW);
    for (int b = 0; b < B_; ++b) {
      gemmBF16(stream, fb16 + (long)b * CHW, 1, N_, 0,
               gb16 + (long)b * CHW, N_, 1, 0, esc, N_, 0,
               nullptr, 1.0f, N_, N_, C_, 1);
      softmax_rows_kernel<<<N_, 256, 0, stream>>>(esc, N_);
      colmean_kernel<<<N_ / 256, 256, 0, stream>>>(esc, N_, N_,
                                                   attens[t] + (long)b * N_,
                                                   1.0f / N_);
    }
  }

  // ---- masks: keep = top-(n-k) in BOTH batches; equality mask + row L1 renorm
  keepmask_kernel<<<16, 256, 0, stream>>>(attenx, N_, KEEP_K, keepx);
  keepmask_kernel<<<16, 256, 0, stream>>>(atteny, N_, KEEP_K, keepy);
  mask_renorm_kernel<<<B_ * N_, 256, 0, stream>>>(corr, keepx, keepy, corrb, N_);

  // ---- h = Wh*y + bh ; residual[b] = h[b] * corr[b]^T  (bf16 GEMM, K=4096)
  gemmF32(stream, h_w, C_, 1, 0, y, N_, 1, CHW, fbuf, N_, CHW,
          h_b, nullptr, 0, 1.0f, 0.0f, C_, N_, C_, B_);
  tobf16_kernel<<<(int)((BCHW + 255) / 256), 256, 0, stream>>>(fbuf, fb16, BCHW);
  gemmBF16(stream, fb16, N_, 1, CHW, corrb, 1, N_, N2, gbuf, N_, CHW,
           nullptr, 1.0f, C_, N_, N_, B_);

  // ---- output = x + Wout*residual + bout -> d_out; then attens
  gemmF32(stream, out_w, C_, 1, 0, gbuf, N_, 1, CHW, out, N_, CHW,
          out_b, x, CHW, 1.0f, 0.0f, C_, N_, C_, B_);
  copy_kernel<<<(B_ * N_ + 255) / 256, 256, 0, stream>>>(attenx, out + BCHW, B_ * N_);
  copy_kernel<<<(B_ * N_ + 255) / 256, 256, 0, stream>>>(atteny, out + BCHW + B_ * N_, B_ * N_);
}
